// QuiPLinear_26912265077058
// MI455X (gfx1250) — matrix operand order repaired and checked by reading, not compiled, not run on hardware
//
#include <hip/hip_runtime.h>

typedef __attribute__((ext_vector_type(16))) _Float16 v16h;
typedef __attribute__((ext_vector_type(8)))  _Float16 v8h;
typedef __attribute__((ext_vector_type(4)))  _Float16 v4h;
typedef __attribute__((ext_vector_type(8)))  float    v8f;
typedef __attribute__((ext_vector_type(4)))  unsigned int v4u;
typedef __attribute__((ext_vector_type(4)))  int      v4i;
typedef __attribute__((ext_vector_type(8)))  int      v8i;

#define N_IN   4096
#define N_OUT  4096
#define BM     128
#define BN     128
#define BK     64
#define LDP    72          // padded LDS row stride in halfs (64 + 8)

#if __has_builtin(__builtin_amdgcn_tensor_load_to_lds) && \
    __has_builtin(__builtin_amdgcn_s_wait_tensorcnt)
#define HAVE_TDM 1
#else
#define HAVE_TDM 0
#endif

// ---------------------------------------------------------------------------
// In-LDS unnormalized FWHT of 4096 f32 values with 256 threads.
// ---------------------------------------------------------------------------
__device__ __forceinline__ void fwht4096(float* __restrict__ s, int tid) {
  for (int h = 1; h < 4096; h <<= 1) {
    __syncthreads();
#pragma unroll 1
    for (int p = tid; p < 2048; p += 256) {
      int i = ((p & ~(h - 1)) << 1) | (p & (h - 1));
      float a = s[i];
      float b = s[i + h];
      s[i]     = a + b;
      s[i + h] = a - b;
    }
  }
  __syncthreads();
}

// ---------------------------------------------------------------------------
// Kernel 1: decode weight row, Hadamard it, fold SU * Wscale/64, emit f16.
// ---------------------------------------------------------------------------
__global__ __launch_bounds__(256) void quip_decode_wh(
    const int* __restrict__ Qidxs, const float* __restrict__ codebook,
    const float* __restrict__ SU, const float* __restrict__ Wscale,
    _Float16* __restrict__ Wh) {
  __shared__ __align__(16) float s[4096];
  const int o = blockIdx.x;
  const int tid = threadIdx.x;

  for (int e = tid; e < 512; e += 256) {
    const int q = Qidxs[o * 512 + e];
    const float4* cb = (const float4*)(codebook + q * 8);
    float4 c0 = cb[0];
    float4 c1 = cb[1];
    ((float4*)s)[e * 2]     = c0;
    ((float4*)s)[e * 2 + 1] = c1;
  }

  fwht4096(s, tid);

  const float scale = Wscale[0] * 0.015625f;  // Wscale / 64
  const int i0 = tid * 16;
#pragma unroll
  for (int v = 0; v < 2; ++v) {
    v8h out;
#pragma unroll
    for (int j = 0; j < 8; ++j) {
      const int i = i0 + v * 8 + j;
      out[j] = (_Float16)(s[i] * SU[i] * scale);
    }
    *(v8h*)(Wh + (size_t)o * N_IN + i0 + v * 8) = out;
  }
}

// ---------------------------------------------------------------------------
// Kernel 2: Z = X(f32->f16) @ Wh(f16)^T, f32 accumulate, WMMA 16x16x32.
// Double-buffered LDS. A staged through VGPRs (needs cvt). B tile fetched by
// the Tensor Data Mover (async, TENSORcnt) when available, else VGPR-staged.
// ---------------------------------------------------------------------------
__global__ __launch_bounds__(256) void quip_gemm_wmma(
    const float* __restrict__ X, const _Float16* __restrict__ Wh,
    float* __restrict__ Z) {
  __shared__ __align__(16) _Float16 sA[2][BM * LDP];
  __shared__ __align__(16) _Float16 sB[2][BN * LDP];

  const int tid    = threadIdx.x;
  const int lane   = tid & 31;
  const int llo    = lane & 15;
  const int lhi    = lane >> 4;
  const int waveId = tid >> 5;
  const int waveM  = waveId & 3;   // 0..3
  const int waveN  = waveId >> 2;  // 0..1

  const int tileM = blockIdx.y * BM;
  const int tileN = blockIdx.x * BN;

  v8f acc[2][4] = {};

  // ---- staging decomposition ----
  const int arow = tid >> 4;        // 16 rows / pass, 8 passes (A)
  const int acol = (tid & 15) * 4;  // 4 f32 / thread / pass
  const float* gA = X + (size_t)(tileM + arow) * N_IN + acol;
  float4 ra[8];

#if !HAVE_TDM
  const int brow = tid >> 2;        // 64 rows / pass, 2 passes (B)
  const int bcol = (tid & 3) * 16;  // 8 halfs (uint4) / thread / pass, 16-half col step
  const _Float16* gB = Wh + (size_t)(tileN + brow) * N_IN + bcol;
  uint4 rb[2];
#endif

  auto load_tile = [&](int k0) {
#pragma unroll
    for (int p = 0; p < 8; ++p)
      ra[p] = *(const float4*)(gA + (size_t)p * 16 * N_IN + k0);
#if !HAVE_TDM
#pragma unroll
    for (int p = 0; p < 2; ++p)
      rb[p] = *(const uint4*)(gB + (size_t)p * 64 * N_IN + k0);
#endif
  };

  auto store_tile = [&](int buf) {
#pragma unroll
    for (int p = 0; p < 8; ++p) {
      v4h h4;
      h4[0] = (_Float16)ra[p].x; h4[1] = (_Float16)ra[p].y;
      h4[2] = (_Float16)ra[p].z; h4[3] = (_Float16)ra[p].w;
      *(v4h*)(&sA[buf][(p * 16 + arow) * LDP + acol]) = h4;
    }
#if !HAVE_TDM
#pragma unroll
    for (int p = 0; p < 2; ++p)
      *(uint4*)(&sB[buf][(p * 64 + brow) * LDP + bcol]) = rb[p];
#endif
  };

#if HAVE_TDM
  // D# group 1 (constant): mask=0, data_size=2B(code 1), pad_enable,
  // pad_interval=32 DWORDs (code 4) -> pad every 64 halfs,
  // pad_amount=4 DWORDs (code 3) -> +8 halfs  ==> LDS row stride 72 halfs.
  v8i g1;
  g1[0] = (1 << 16) | (1 << 20) | (4 << 22) | (3 << 25);
  g1[1] = (int)((N_IN & 0xFFFF) << 16);          // tensor_dim0[15:0] @ bits63:48
  g1[2] = (int)(((N_IN >> 16) & 0xFFFF) |        // tensor_dim0[31:16]
                ((N_IN & 0xFFFF) << 16));        // tensor_dim1[15:0]
  g1[3] = (int)(((N_IN >> 16) & 0xFFFF) |        // tensor_dim1[31:16]
                (BK << 16));                     // tile_dim0 = 64
  g1[4] = BN;                                    // tile_dim1 = 128 (tile_dim2=0)
  g1[5] = N_IN;                                  // tensor_dim0_stride[31:0]
  g1[6] = 0;
  g1[7] = 0;
  const v4i gz = {0, 0, 0, 0};

  auto tdm_issue_b = [&](int k0, int buf) {
    const unsigned long long ga =
        (unsigned long long)(const void*)(Wh + (size_t)tileN * N_IN + k0);
    const unsigned int lds = (unsigned int)(size_t)&sB[buf][0];
    v4u g0;
    g0[0] = 1u;                                   // count=1 (valid), no gather
    g0[1] = lds;                                  // lds_addr
    g0[2] = (unsigned int)ga;                     // global_addr[31:0]
    g0[3] = (unsigned int)(ga >> 32) | 0x80000000u;  // addr[56:32] | type=2
#if __clang_major__ >= 23
    const v8i gz8 = {0, 0, 0, 0, 0, 0, 0, 0};
    __builtin_amdgcn_tensor_load_to_lds(g0, g1, gz, gz, gz8, 0);
#else
    __builtin_amdgcn_tensor_load_to_lds(g0, g1, gz, gz, 0);
#endif
  };
#endif

  // ---------------- software pipeline: prologue ----------------
  load_tile(0);
#if HAVE_TDM
  if (waveId == 0) tdm_issue_b(0, 0);
#endif
  store_tile(0);
#if HAVE_TDM
  if (waveId == 0) __builtin_amdgcn_s_wait_tensorcnt(0);
#endif
  __syncthreads();

  int buf = 0;
  for (int k0 = 0; k0 < N_IN; k0 += BK) {
    const bool more = (k0 + BK) < N_IN;
    if (more) {
      load_tile(k0 + BK);                  // global loads for next tile
#if HAVE_TDM
      if (waveId == 0) tdm_issue_b(k0 + BK, buf ^ 1);
#endif
    }

    // ---------------- compute current buffer: 2 K-steps of 32 ----------------
    const _Float16* bA = sA[buf];
    const _Float16* bB = sB[buf];
#pragma unroll
    for (int kk = 0; kk < BK; kk += 32) {
      v16h af[2], bf[4];
#pragma unroll
      for (int mf = 0; mf < 2; ++mf) {
        // A 16x32 f16: lanes<16 -> K[kk..+7],K[kk+16..+23]; lanes>=16 -> +8
        const _Float16* pa = bA + (waveM * 32 + mf * 16 + llo) * LDP + kk + lhi * 8;
        union { v16h v; v8h h[2]; } u;
        u.h[0] = *(const v8h*)(pa);
        u.h[1] = *(const v8h*)(pa + 16);
        af[mf] = u.v;
      }
#pragma unroll
      for (int nf = 0; nf < 4; ++nf) {
        // B 32x16 f16: lanes<16 -> K[kk..+7],K[kk+16..+23]; lanes>=16 -> +8
        const _Float16* pb = bB + (waveN * 64 + nf * 16 + llo) * LDP + kk + lhi * 8;
        union { v16h v; v8h h[2]; } u;
        u.h[0] = *(const v8h*)(pb);
        u.h[1] = *(const v8h*)(pb + 16);
        bf[nf] = u.v;
      }
#pragma unroll
      for (int mf = 0; mf < 2; ++mf)
#pragma unroll
        for (int nf = 0; nf < 4; ++nf)
          acc[mf][nf] = __builtin_amdgcn_wmma_f32_16x16x32_f16(
              false, af[mf], false, bf[nf], (short)0, acc[mf][nf],
              false, false);
    }

    if (more) {
      store_tile(buf ^ 1);                 // cvt + LDS stores for next tile
#if HAVE_TDM
      if (waveId == 0) __builtin_amdgcn_s_wait_tensorcnt(0);
#endif
      __syncthreads();                     // publish next buffer, fence skew
      buf ^= 1;
    }
  }

  // ---- write out: C/D layout: VGPR r, lanes<16 -> M=r, lanes>=16 -> M=r+8 ----
#pragma unroll
  for (int mf = 0; mf < 2; ++mf) {
#pragma unroll
    for (int nf = 0; nf < 4; ++nf) {
      const int col = tileN + waveN * 64 + nf * 16 + llo;
#pragma unroll
      for (int r = 0; r < 8; ++r) {
        const int row = tileM + waveM * 32 + mf * 16 + r + lhi * 8;
        Z[(size_t)row * N_OUT + col] = acc[mf][nf][r];
      }
    }
  }
}

// ---------------------------------------------------------------------------
// Kernel 3: in-place FWHT of each z row (len 4096) + SV/64 scaling.
// ---------------------------------------------------------------------------
__global__ __launch_bounds__(256) void quip_fwht_out(
    float* __restrict__ Z, const float* __restrict__ SV) {
  __shared__ __align__(16) float s[4096];
  const int t = blockIdx.x;
  const int tid = threadIdx.x;
  float4* row = (float4*)(Z + (size_t)t * N_OUT);

#pragma unroll
  for (int v = 0; v < 4; ++v)
    ((float4*)s)[v * 256 + tid] = row[v * 256 + tid];

  fwht4096(s, tid);

#pragma unroll
  for (int v = 0; v < 4; ++v) {
    const int idx4 = v * 256 + tid;
    const int i = idx4 * 4;
    float4 val = ((float4*)s)[idx4];
    val.x *= 0.015625f * SV[i + 0];
    val.y *= 0.015625f * SV[i + 1];
    val.z *= 0.015625f * SV[i + 2];
    val.w *= 0.015625f * SV[i + 3];
    row[idx4] = val;
  }
}

// ---------------------------------------------------------------------------
extern "C" void kernel_launch(void* const* d_in, const int* in_sizes, int n_in,
                              void* d_out, int out_size, void* d_ws,
                              size_t ws_size, hipStream_t stream) {
  const float* input    = (const float*)d_in[0];  // [B,S,IN] f32
  const int*   Qidxs    = (const int*)  d_in[1];  // [OUT, IN/8] i32
  const float* codebook = (const float*)d_in[2];  // [256, 8] f32
  const float* SU       = (const float*)d_in[3];  // [IN] f32
  const float* SV       = (const float*)d_in[4];  // [OUT] f32
  const float* Wscale   = (const float*)d_in[5];  // scalar f32
  float* out = (float*)d_out;                     // [T, OUT] f32

  const int T = in_sizes[0] / N_IN;               // 8192 tokens
  _Float16* Wh = (_Float16*)d_ws;                 // needs OUT*IN*2 = 32 MiB

  quip_decode_wh<<<N_OUT, 256, 0, stream>>>(Qidxs, codebook, SU, Wscale, Wh);

  dim3 grid(N_OUT / BN, T / BM);
  quip_gemm_wmma<<<grid, 256, 0, stream>>>(input, Wh, out);

  quip_fwht_out<<<T, 256, 0, stream>>>(out, SV);
}
